// GNN_pairs_29635274342454
// MI455X (gfx1250) — matrix operand order repaired and checked by reading, not compiled, not run on hardware
//
#include <hip/hip_runtime.h>

// ---------------------------------------------------------------------------
// GATv2 GNN forward for MI455X (gfx1250, wave32, WMMA).
//
// Dense GEMMs (x@Wl, x@Wr, h@Wt, MLP) run on v_wmma_f32_16x16x32_bf16 with
// fp32 accumulation. Weights are pre-swizzled to a B-fragment layout (k = 16h + s) in
// global memory so each lane loads its fragment as 2x b128; the A strip is
// staged to LDS once per block and fragments load as 2x ds_load_b128.
// Edge attention (ee = ea@We with K<=6, leaky-relu, att dot, segment softmax
// over dst, alpha-weighted scatter) is fused into wave-per-edge kernels with
// fp32 atomics; segment max uses the int-punned float atomic-max trick
// (self-loops guarantee every node participates).
//
// d_in layout assumption (jax pytree flatten, dict keys sorted):
//   0 x(20000x41) 1 ei1(2x80000) 2 ea1(80000x6) 3 ei2(2x160000)
//   4 ea2(160000x3) 5 batch(20000) 6 num_graphs
//   params: dense: 7 w0(64x256) 8 b0 9 w1(256x128) 10 b1 11 w2(128x64) 12 b2
//   g1 layers l=0..2 at base 13+9l: We,Wl,Wr,Wt,att,beta,bias,bt,gamma
//   g2 layers at base 40+9l (same order)
//   67 gate1_w(64x1) 68 gate1_b 69 gate2_w 70 gate2_b 71 out_w(64x1) 72 out_b
// ---------------------------------------------------------------------------

typedef __attribute__((ext_vector_type(16))) __bf16 v16bf;
typedef __attribute__((ext_vector_type(8)))  __bf16 v8bf;
typedef __attribute__((ext_vector_type(8)))  float  v8f;

#define NND    20000
#define FIND   41
#define EMBD   64
#define NGRD   800
#define NE1D   80000
#define NED1D  6
#define NE2D   160000
#define NED2D  3
#define NEGINF (-3.402823466e38f)

__device__ __forceinline__ unsigned short f2bf(float f) {
  unsigned int u = __float_as_uint(f);
  unsigned int r = u + 0x7fffu + ((u >> 16) & 1u);
  if ((u & 0x7f800000u) == 0x7f800000u) r = u;   // inf/nan passthrough
  return (unsigned short)(r >> 16);
}

__device__ __forceinline__ void atomicMaxFloat(float* addr, float val) {
  if (val >= 0.0f) atomicMax((int*)addr, __float_as_int(val));
  else             atomicMin((unsigned int*)addr, __float_as_uint(val));
}

// ----------------------------- utility kernels -----------------------------

__global__ void k_fill(float* __restrict__ p, float v, long n) {
  long t = (long)blockIdx.x * blockDim.x + threadIdx.x;
  if (t < n) p[t] = v;
}

// A (MxKin f32, optional +bias per col) -> bf16 MxKpad row-major, zero padded
__global__ void k_conv_a(const float* __restrict__ X, const float* __restrict__ bias,
                         unsigned short* __restrict__ out, int M, int Kin, int Kpad) {
  long t = (long)blockIdx.x * blockDim.x + threadIdx.x;
  if (t >= (long)M * Kpad) return;
  int  k = (int)(t % Kpad);
  long r = t / Kpad;
  float v = 0.0f;
  if (k < Kin) { v = X[r * Kin + k]; if (bias) v += bias[k]; }
  out[t] = f2bf(v);
}

// W (KinxNout f32) -> bf16 in WMMA-B fragment layout:
//   flat index = ((chunk*NT + ct)*32 + lane)*16 + elem,  NT = Nout/16
//   k = chunk*32 + (lane>=16 ? 16 : 0) + elem, c = ct*16 + (lane&15)
__global__ void k_conv_w_sw(const float* __restrict__ W, unsigned short* __restrict__ out,
                            int Kin, int Kpad, int Nout) {
  int t = blockIdx.x * blockDim.x + threadIdx.x;
  if (t >= Kpad * Nout) return;
  int elem = t & 15;
  int lane = (t >> 4) & 31;
  int rest = t >> 9;                    // chunk*NT + ct
  int NT = Nout >> 4;
  int chunk = rest / NT, ct = rest - chunk * NT;
  int k = chunk * 32 + ((lane >> 4) << 4) + elem;
  int c = ct * 16 + (lane & 15);
  out[t] = (k < Kin) ? f2bf(W[(long)k * Nout + c]) : (unsigned short)0;
}

// ------------------------------- WMMA GEMM ---------------------------------
// 256 threads = 8 waves. NT = NOUT/16 column tiles; WPS = min(NT,8) waves per
// row strip; S = 8/WPS strips per block (keeps all waves busy for NOUT=64);
// CTPW = NT/WPS column tiles per wave. Requires M % (16*S) == 0.
// act: 0 = none, 1 = relu.

template <int NOUT, int KPAD>
__global__ __launch_bounds__(256) void k_gemm(
    const unsigned short* __restrict__ A, const unsigned short* __restrict__ Wsw,
    const float* __restrict__ bias, float* __restrict__ C, int M, int act) {
  constexpr int NT   = NOUT / 16;
  constexpr int WPS  = (NT < 8) ? NT : 8;
  constexpr int S    = 8 / WPS;
  constexpr int CTPW = NT / WPS;
  __shared__ unsigned short sA[S * 16 * KPAD];

  const int tid = threadIdx.x;
  const long rowBase0 = (long)blockIdx.x * (16 * S);

  // flat b128 copy of S*16 contiguous bf16 rows into LDS
  {
    const uint4* g = (const uint4*)(A + rowBase0 * KPAD);
    uint4* l = (uint4*)sA;
    for (int i = tid; i < (S * 16 * KPAD) / 8; i += 256) l[i] = g[i];
  }
  __syncthreads();

  const int wave = tid >> 5, lane = tid & 31;
  const int strip = wave / WPS, wv = wave - strip * WPS;
  const int rA  = lane & 15;
  const int hiA = lane >> 4;            // 0: lanes 0-15, 1: lanes 16-31

  v8f acc[CTPW];
#pragma unroll
  for (int t = 0; t < CTPW; ++t)
#pragma unroll
    for (int v = 0; v < 8; ++v) acc[t][v] = 0.0f;

  const unsigned short* sArow = &sA[(strip * 16 + rA) * KPAD + hiA * 16];

#pragma unroll
  for (int kc = 0; kc < KPAD; kc += 32) {
    // A fragment (contiguous k = 16*h + s, matching B packing): 32B run per lane
    v8bf alo = *(const v8bf*)(sArow + kc);
    v8bf ahi = *(const v8bf*)(sArow + kc + 8);
    v16bf afrag = __builtin_shufflevector(alo, ahi, 0, 1, 2, 3, 4, 5, 6, 7,
                                          8, 9, 10, 11, 12, 13, 14, 15);
#pragma unroll
    for (int t = 0; t < CTPW; ++t) {
      const int ct = wv + t * WPS;
      const unsigned short* wp = Wsw + ((long)((kc >> 5) * NT + ct) << 9) + lane * 16;
      v8bf blo = *(const v8bf*)wp;
      v8bf bhi = *(const v8bf*)(wp + 8);
      v16bf bfrag = __builtin_shufflevector(blo, bhi, 0, 1, 2, 3, 4, 5, 6, 7,
                                            8, 9, 10, 11, 12, 13, 14, 15);
      acc[t] = __builtin_amdgcn_wmma_f32_16x16x32_bf16(
          false, afrag, false, bfrag, (short)0, acc[t], false, false);
    }
  }

#pragma unroll
  for (int t = 0; t < CTPW; ++t) {
    const int ct = wv + t * WPS;
    const int col = ct * 16 + rA;
    const float b = bias ? bias[col] : 0.0f;
#pragma unroll
    for (int v = 0; v < 8; ++v) {
      long row = rowBase0 + strip * 16 + v + hiA * 8;   // ISA C layout
      float cv = acc[t][v] + b;
      if (act) cv = fmaxf(cv, 0.0f);
      C[row * NOUT + col] = cv;
    }
  }
}

// ----------------------------- edge kernels --------------------------------

__global__ void k_edge_deg(const int* __restrict__ dst, const float* __restrict__ ea,
                           float* __restrict__ loop_sum, float* __restrict__ deg,
                           int E, int ED) {
  int e = blockIdx.x * blockDim.x + threadIdx.x;
  if (e >= E) return;
  int d = dst[e];
  atomicAdd(&deg[d], 1.0f);
  __builtin_prefetch(&ea[(long)e * ED], 0, 1);
  for (int k = 0; k < ED; ++k) atomicAdd(&loop_sum[(long)d * 8 + k], ea[(long)e * ED + k]);
}

__global__ void k_loop_final(float* __restrict__ loop, const float* __restrict__ deg,
                             int M, int ED) {
  int n = blockIdx.x * blockDim.x + threadIdx.x;
  if (n >= M) return;
  float dg = fmaxf(deg[n], 1.0f);
  for (int k = 0; k < 8; ++k) {
    float v = (k < ED) ? loop[(long)n * 8 + k] / dg : 0.0f;
    loop[(long)n * 8 + k] = v;
  }
}

// One wave per edge (8/block). lane covers 8 channels; head = lane>>3.
__global__ __launch_bounds__(256) void k_edge_logit(
    const int* __restrict__ src, const int* __restrict__ dst,
    const float* __restrict__ ea, int ED, int E, int Etot,
    const float* __restrict__ We, const float* __restrict__ att,
    const float* __restrict__ loop_attr,
    const float* __restrict__ xl, const float* __restrict__ xr,
    float* __restrict__ logit, float* __restrict__ lmax) {
  __shared__ float sWe[8 * 256];
  __shared__ float sAtt[256];
  const int tid = threadIdx.x;
  for (int i = tid; i < 8 * 256; i += 256) {
    int k = i >> 8;
    sWe[i] = (k < ED) ? We[(long)k * 256 + (i & 255)] : 0.0f;
  }
  sAtt[tid] = att[tid];
  __syncthreads();

  const int wave = tid >> 5, lane = tid & 31;
  const int e = blockIdx.x * 8 + wave;
  if (e >= Etot) return;

  int s, d;
  float eav[8];
  if (e < E) {
    s = src[e]; d = dst[e];
#pragma unroll
    for (int k = 0; k < 8; ++k) eav[k] = (k < ED) ? ea[(long)e * ED + k] : 0.0f;
  } else {
    s = d = e - E;
#pragma unroll
    for (int k = 0; k < 8; ++k) eav[k] = loop_attr[(long)(e - E) * 8 + k];
  }

  union { float4 v4[2]; float f[8]; } uxl, uxr;
  uxl.v4[0] = ((const float4*)(xl + (long)s * 256 + lane * 8))[0];
  uxl.v4[1] = ((const float4*)(xl + (long)s * 256 + lane * 8))[1];
  uxr.v4[0] = ((const float4*)(xr + (long)d * 256 + lane * 8))[0];
  uxr.v4[1] = ((const float4*)(xr + (long)d * 256 + lane * 8))[1];

  float part = 0.0f;
#pragma unroll
  for (int j = 0; j < 8; ++j) {
    int c = lane * 8 + j;
    float ee = 0.0f;
#pragma unroll
    for (int k = 0; k < 8; ++k) ee += eav[k] * sWe[k * 256 + c];
    float v = uxl.f[j] + uxr.f[j] + ee;
    v = (v > 0.0f) ? v : 0.2f * v;        // leaky relu, slope 0.2
    part += v * sAtt[c];
  }
  part += __shfl_xor(part, 1);
  part += __shfl_xor(part, 2);
  part += __shfl_xor(part, 4);
  if ((lane & 7) == 0) {
    int head = lane >> 3;
    logit[(long)e * 4 + head] = part;
    atomicMaxFloat(&lmax[(long)d * 4 + head], part);
  }
}

__global__ void k_edge_ex(const int* __restrict__ dst, int E, int Etot,
                          const float* __restrict__ lmax,
                          float* __restrict__ lx, float* __restrict__ den) {
  int t = blockIdx.x * blockDim.x + threadIdx.x;
  if (t >= Etot * 4) return;
  int e = t >> 2, hd = t & 3;
  int d = (e < E) ? dst[e] : (e - E);
  float x = __expf(lx[t] - lmax[(long)d * 4 + hd]);
  lx[t] = x;
  atomicAdd(&den[(long)d * 4 + hd], x);
}

__global__ __launch_bounds__(256) void k_edge_scatter(
    const int* __restrict__ src, const int* __restrict__ dst, int E, int Etot,
    const float* __restrict__ ex, const float* __restrict__ den,
    const float* __restrict__ xl, float* __restrict__ acc) {
  int wave = threadIdx.x >> 5, lane = threadIdx.x & 31;
  int e = blockIdx.x * 8 + wave;
  if (e >= Etot) return;
  int s, d;
  if (e < E) { s = src[e]; d = dst[e]; } else { s = d = e - E; }

  union { float4 v4[2]; float f[8]; } uxl;
  uxl.v4[0] = ((const float4*)(xl + (long)s * 256 + lane * 8))[0];
  uxl.v4[1] = ((const float4*)(xl + (long)s * 256 + lane * 8))[1];

  float a0 = ex[(long)e * 4 + ((lane * 8) >> 6)] / den[(long)d * 4 + ((lane * 8) >> 6)];
#pragma unroll
  for (int j = 0; j < 8; ++j) {
    int c = lane * 8 + j;
    atomicAdd(&acc[(long)d * 256 + c], a0 * uxl.f[j]);
  }
}

// ------------------------------ batch norm ---------------------------------

__global__ __launch_bounds__(256) void k_bn_stats(const float* __restrict__ h,
                                                  float* __restrict__ stat, int M) {
  __shared__ float ssum[256], ssq[256];
  int t = threadIdx.x;
  int c = t & 63, rsub = t >> 6;
  long rowBase = (long)blockIdx.x * 64;
  float s = 0.0f, q = 0.0f;
  for (int r = rsub; r < 64; r += 4) {
    long row = rowBase + r;
    if (row < M) { float v = h[row * 64 + c]; s += v; q += v * v; }
  }
  ssum[t] = s; ssq[t] = q;
  __syncthreads();
  if (t < 64) {
    s = ssum[t] + ssum[t + 64] + ssum[t + 128] + ssum[t + 192];
    q = ssq[t]  + ssq[t + 64]  + ssq[t + 128]  + ssq[t + 192];
    atomicAdd(&stat[c], s);
    atomicAdd(&stat[64 + c], q);
  }
}

__global__ void k_bn_apply(float* __restrict__ h, const float* __restrict__ stat,
                           const float* __restrict__ gamma, const float* __restrict__ beta,
                           int M) {
  long t = (long)blockIdx.x * blockDim.x + threadIdx.x;
  if (t >= (long)M * 64) return;
  int c = (int)(t & 63);
  float mu  = stat[c] / (float)M;
  float var = stat[64 + c] / (float)M - mu * mu;
  h[t] = gamma[c] * (h[t] - mu) * rsqrtf(var + 1e-5f) + beta[c];
}

// -------------------------------- pooling ----------------------------------

__global__ __launch_bounds__(256) void k_gate(const float* __restrict__ h,
                                              const float* __restrict__ gw,
                                              const float* __restrict__ gb,
                                              const int* __restrict__ batch,
                                              float* __restrict__ gval,
                                              float* __restrict__ gmax, int M) {
  int wave = threadIdx.x >> 5, lane = threadIdx.x & 31;
  int n = blockIdx.x * 8 + wave;
  if (n >= M) return;
  float s = h[(long)n * 64 + lane] * gw[lane] + h[(long)n * 64 + 32 + lane] * gw[32 + lane];
#pragma unroll
  for (int m = 16; m; m >>= 1) s += __shfl_xor(s, m);
  if (lane == 0) {
    s += gb[0];
    gval[n] = s;
    atomicMaxFloat(&gmax[batch[n]], s);
  }
}

__global__ void k_gate_ex(const int* __restrict__ batch, float* __restrict__ gval,
                          const float* __restrict__ gmax, float* __restrict__ gden, int M) {
  int n = blockIdx.x * blockDim.x + threadIdx.x;
  if (n >= M) return;
  int b = batch[n];
  float x = __expf(gval[n] - gmax[b]);
  gval[n] = x;
  atomicAdd(&gden[b], x);
}

__global__ void k_pool_scatter(const float* __restrict__ h, const float* __restrict__ gval,
                               const float* __restrict__ gden, const int* __restrict__ batch,
                               float* __restrict__ pool, int M) {
  long t = (long)blockIdx.x * blockDim.x + threadIdx.x;
  if (t >= (long)M * 64) return;
  int n = (int)(t >> 6), c = (int)(t & 63);
  int b = batch[n];
  float w = gval[n] / gden[b];
  atomicAdd(&pool[(long)b * 64 + c], w * h[t]);
}

__global__ void k_sub(const float* __restrict__ a, const float* __restrict__ b,
                      float* __restrict__ z, int n) {
  int t = blockIdx.x * blockDim.x + threadIdx.x;
  if (t < n) z[t] = a[t] - b[t];
}

__global__ __launch_bounds__(256) void k_final(const float* __restrict__ z,
                                               const float* __restrict__ wo,
                                               const float* __restrict__ bo,
                                               float* __restrict__ out, int M) {
  int wave = threadIdx.x >> 5, lane = threadIdx.x & 31;
  int g = blockIdx.x * 8 + wave;
  if (g >= M) return;
  float s = z[(long)g * 64 + lane] * wo[lane] + z[(long)g * 64 + 32 + lane] * wo[32 + lane];
#pragma unroll
  for (int m = 16; m; m >>= 1) s += __shfl_xor(s, m);
  if (lane == 0) out[g] = s + bo[0];
}

// ------------------------------- host side ---------------------------------

static inline unsigned nblk(long n, int b) { return (unsigned)((n + b - 1) / b); }

extern "C" void kernel_launch(void* const* d_in, const int* in_sizes, int n_in,
                              void* d_out, int out_size, void* d_ws, size_t ws_size,
                              hipStream_t stream) {
  (void)in_sizes; (void)n_in; (void)out_size; (void)ws_size;

  const float* x     = (const float*)d_in[0];
  const int*   ei1   = (const int*)d_in[1];
  const float* ea1   = (const float*)d_in[2];
  const int*   ei2   = (const int*)d_in[3];
  const float* ea2   = (const float*)d_in[4];
  const int*   batch = (const int*)d_in[5];

  // workspace carve-up
  char* ws = (char*)d_ws;
  size_t off = 0;
  auto alloc = [&](size_t bytes) -> void* {
    off = (off + 255) & ~(size_t)255;
    void* p = ws + off;
    off += bytes;
    return p;
  };
  float* XL    = (float*)alloc((size_t)NND * 256 * 4);
  float* XR    = (float*)alloc((size_t)NND * 256 * 4);
  float* ACC   = (float*)alloc((size_t)NND * 256 * 4);
  float* H0    = (float*)alloc((size_t)NND * 64 * 4);
  float* H1    = (float*)alloc((size_t)NND * 64 * 4);
  float* LOGIT = (float*)alloc((size_t)(NE2D + NND) * 4 * 4);
  float* LMAX  = (float*)alloc((size_t)NND * 4 * 4);
  float* DEN   = (float*)alloc((size_t)NND * 4 * 4);
  float* LOOP  = (float*)alloc((size_t)NND * 8 * 4);
  float* DEG   = (float*)alloc((size_t)NND * 4);
  float* STAT  = (float*)alloc(128 * 4);
  float* POOL1 = (float*)alloc((size_t)NGRD * 64 * 4);
  float* POOL2 = (float*)alloc((size_t)NGRD * 64 * 4);
  float* GVAL  = (float*)alloc((size_t)NND * 4);
  float* GMAX  = (float*)alloc((size_t)NGRD * 4);
  float* GDEN  = (float*)alloc((size_t)NGRD * 4);
  float* Z0    = (float*)alloc((size_t)NGRD * 64 * 4);
  float* Z1    = (float*)alloc((size_t)NGRD * 256 * 4);
  float* Z2    = (float*)alloc((size_t)NGRD * 128 * 4);
  float* Z3    = (float*)alloc((size_t)NGRD * 64 * 4);
  unsigned short* ABF = (unsigned short*)alloc((size_t)NND * 256 * 2);
  unsigned short* WBF = (unsigned short*)alloc((size_t)256 * 256 * 2);
  unsigned short* ZBF = (unsigned short*)alloc((size_t)NGRD * 256 * 2);

  auto fill = [&](float* p, float v, long n) {
    k_fill<<<nblk(n, 256), 256, 0, stream>>>(p, v, n);
  };
  // GEMM dispatch: shapes are (Nout,Kpad) in {(256,64),(64,256),(128,256),(64,128)}
  auto gemm = [&](const unsigned short* A, const unsigned short* Wb, const float* bias,
                  float* C, int M, int Kpad, int Nout, int act) {
    int S = (Nout == 64) ? 2 : 1;            // strips per block (see k_gemm)
    dim3 g((unsigned)(M / (16 * S))), b(256);
    if      (Nout == 256 && Kpad == 64)  k_gemm<256, 64><<<g, b, 0, stream>>>(A, Wb, bias, C, M, act);
    else if (Nout == 64  && Kpad == 256) k_gemm<64, 256><<<g, b, 0, stream>>>(A, Wb, bias, C, M, act);
    else if (Nout == 128 && Kpad == 256) k_gemm<128, 256><<<g, b, 0, stream>>>(A, Wb, bias, C, M, act);
    else if (Nout == 64  && Kpad == 128) k_gemm<64, 128><<<g, b, 0, stream>>>(A, Wb, bias, C, M, act);
  };

  auto run_branch = [&](const int* ei, const float* ea, int E, int ED, int pbase,
                        float* pool_out, const float* gate_w, const float* gate_b) {
    const float* hin = x;
    int Kin = FIND;
    float* hbuf[2] = {H0, H1};
    const float* hout = nullptr;
    for (int l = 0; l < 3; ++l) {
      const float* We    = (const float*)d_in[pbase + 9 * l + 0];
      const float* Wl    = (const float*)d_in[pbase + 9 * l + 1];
      const float* Wr    = (const float*)d_in[pbase + 9 * l + 2];
      const float* Wt    = (const float*)d_in[pbase + 9 * l + 3];
      const float* att   = (const float*)d_in[pbase + 9 * l + 4];
      const float* beta  = (const float*)d_in[pbase + 9 * l + 5];
      const float* bias  = (const float*)d_in[pbase + 9 * l + 6];
      const float* bt    = (const float*)d_in[pbase + 9 * l + 7];
      const float* gamma = (const float*)d_in[pbase + 9 * l + 8];
      const int* src = ei;
      const int* dst = ei + E;
      const int Etot = E + NND;

      // self-loop edge attr = mean of incoming edge attrs
      fill(DEG, 0.0f, NND);
      fill(LOOP, 0.0f, (long)NND * 8);
      k_edge_deg<<<nblk(E, 256), 256, 0, stream>>>(dst, ea, LOOP, DEG, E, ED);
      k_loop_final<<<nblk(NND, 256), 256, 0, stream>>>(LOOP, DEG, NND, ED);

      // xl = h @ Wl, xr = h @ Wr (WMMA)
      int Kpad = (Kin + 31) & ~31;
      k_conv_a<<<nblk((long)NND * Kpad, 256), 256, 0, stream>>>(hin, nullptr, ABF, NND, Kin, Kpad);
      k_conv_w_sw<<<nblk(Kpad * 256, 256), 256, 0, stream>>>(Wl, WBF, Kin, Kpad, 256);
      gemm(ABF, WBF, nullptr, XL, NND, Kpad, 256, 0);
      k_conv_w_sw<<<nblk(Kpad * 256, 256), 256, 0, stream>>>(Wr, WBF, Kin, Kpad, 256);
      gemm(ABF, WBF, nullptr, XR, NND, Kpad, 256, 0);

      // attention logits + segment softmax over dst
      fill(LMAX, NEGINF, (long)NND * 4);
      fill(DEN, 0.0f, (long)NND * 4);
      k_edge_logit<<<nblk(Etot, 8), 256, 0, stream>>>(src, dst, ea, ED, E, Etot,
                                                      We, att, LOOP, XL, XR, LOGIT, LMAX);
      k_edge_ex<<<nblk((long)Etot * 4, 256), 256, 0, stream>>>(dst, E, Etot, LMAX, LOGIT, DEN);
      fill(ACC, 0.0f, (long)NND * 256);
      k_edge_scatter<<<nblk(Etot, 8), 256, 0, stream>>>(src, dst, E, Etot, LOGIT, DEN, XL, ACC);

      // h = relu((acc + bias) @ Wt + bt), then BatchNorm
      k_conv_a<<<nblk((long)NND * 256, 256), 256, 0, stream>>>(ACC, bias, ABF, NND, 256, 256);
      k_conv_w_sw<<<nblk(256 * 64, 256), 256, 0, stream>>>(Wt, WBF, 256, 256, 64);
      float* hn = hbuf[l & 1];
      gemm(ABF, WBF, bt, hn, NND, 256, 64, 1);
      fill(STAT, 0.0f, 128);
      k_bn_stats<<<nblk(NND, 64), 256, 0, stream>>>(hn, STAT, NND);
      k_bn_apply<<<nblk((long)NND * 64, 256), 256, 0, stream>>>(hn, STAT, gamma, beta, NND);
      hin = hn;
      hout = hn;
      Kin = EMBD;
    }
    // attentional aggregation pooling
    fill(GMAX, NEGINF, NGRD);
    fill(GDEN, 0.0f, NGRD);
    fill(pool_out, 0.0f, (long)NGRD * 64);
    k_gate<<<nblk(NND, 8), 256, 0, stream>>>(hout, gate_w, gate_b, batch, GVAL, GMAX, NND);
    k_gate_ex<<<nblk(NND, 256), 256, 0, stream>>>(batch, GVAL, GMAX, GDEN, NND);
    k_pool_scatter<<<nblk((long)NND * 64, 256), 256, 0, stream>>>(hout, GVAL, GDEN, batch,
                                                                  pool_out, NND);
  };

  run_branch(ei1, ea1, NE1D, NED1D, 13, POOL1, (const float*)d_in[67], (const float*)d_in[68]);
  run_branch(ei2, ea2, NE2D, NED2D, 40, POOL2, (const float*)d_in[69], (const float*)d_in[70]);

  // z = pool2 - pool1, then MLP head (WMMA) 64 -> 256 -> 128 -> 64 -> 1
  k_sub<<<nblk(NGRD * 64, 256), 256, 0, stream>>>(POOL2, POOL1, Z0, NGRD * 64);

  k_conv_a<<<nblk((long)NGRD * 64, 256), 256, 0, stream>>>(Z0, nullptr, ZBF, NGRD, 64, 64);
  k_conv_w_sw<<<nblk(64 * 256, 256), 256, 0, stream>>>((const float*)d_in[7], WBF, 64, 64, 256);
  gemm(ZBF, WBF, (const float*)d_in[8], Z1, NGRD, 64, 256, 1);

  k_conv_a<<<nblk((long)NGRD * 256, 256), 256, 0, stream>>>(Z1, nullptr, ZBF, NGRD, 256, 256);
  k_conv_w_sw<<<nblk(256 * 128, 256), 256, 0, stream>>>((const float*)d_in[9], WBF, 256, 256, 128);
  gemm(ZBF, WBF, (const float*)d_in[10], Z2, NGRD, 256, 128, 1);

  k_conv_a<<<nblk((long)NGRD * 128, 256), 256, 0, stream>>>(Z2, nullptr, ZBF, NGRD, 128, 128);
  k_conv_w_sw<<<nblk(128 * 64, 256), 256, 0, stream>>>((const float*)d_in[11], WBF, 128, 128, 64);
  gemm(ZBF, WBF, (const float*)d_in[12], Z3, NGRD, 128, 64, 1);

  k_final<<<nblk(NGRD, 8), 256, 0, stream>>>(Z3, (const float*)d_in[71],
                                             (const float*)d_in[72], (float*)d_out, NGRD);
}
